// PlacementExpert_66941360275554
// MI455X (gfx1250) — matrix operand order repaired and checked by reading, not compiled, not run on hardware
//
#include <hip/hip_runtime.h>
#include <math.h>

// ---------------------------------------------------------------------------
// Types for CDNA5 WMMA
// ---------------------------------------------------------------------------
typedef __attribute__((ext_vector_type(16))) __bf16 v16bf;
typedef __attribute__((ext_vector_type(8)))  float  v8f;

union Frag16 {
    uint4 u[2];   // 32 bytes
    v16bf v;
};

// ---------------------------------------------------------------------------
// CDNA5 async global->LDS copy (ASYNCcnt path) + wait
// ---------------------------------------------------------------------------
static __device__ __forceinline__ void async_copy_b128(unsigned ldsOff, const void* gptr) {
    asm volatile("global_load_async_to_lds_b128 %0, %1, off"
                 :: "v"(ldsOff), "v"(gptr) : "memory");
}
static __device__ __forceinline__ void wait_async0() {
    asm volatile("s_wait_asynccnt 0x0" ::: "memory");
}
static __device__ __forceinline__ unsigned lds_off(const void* p) {
    return (unsigned)(unsigned long long)p;   // LDS aperture: offset in addr[31:0]
}

// ---------------------------------------------------------------------------
// bf16 helpers (RNE)
// ---------------------------------------------------------------------------
static __device__ __forceinline__ unsigned short f2bf(float f) {
    union { float f; unsigned u; } x; x.f = f;
    unsigned r = x.u + 0x7FFFu + ((x.u >> 16) & 1u);
    return (unsigned short)(r >> 16);
}
static __device__ __forceinline__ float bf2f(unsigned short s) {
    union { unsigned u; float f; } x; x.u = ((unsigned)s) << 16;
    return x.f;
}
static __device__ __forceinline__ float gelu_erf(float x) {
    return 0.5f * x * (1.0f + erff(x * 0.70710678118654752f));
}
static __device__ __forceinline__ float sigmoidf_(float x) {
    return 1.0f / (1.0f + expf(-x));
}

// ---------------------------------------------------------------------------
// h = x + task_token (broadcast over rows)
// ---------------------------------------------------------------------------
__global__ void init_h_k(const float* __restrict__ x, const float* __restrict__ token,
                         float* __restrict__ h, int total, int D) {
    int i = blockIdx.x * blockDim.x + threadIdx.x;
    if (i < total) h[i] = x[i] + token[i % D];
}

// ---------------------------------------------------------------------------
// Weight convert + transpose: W[K,N] f32 -> Wt[N,K] bf16.  32x32 tiles.
// ---------------------------------------------------------------------------
__global__ void convT_bf16_k(const float* __restrict__ W, unsigned short* __restrict__ Wt,
                             int K, int N) {
    __shared__ float t[32][33];
    const int n0 = blockIdx.x * 32, k0 = blockIdx.y * 32;
    const int tx = threadIdx.x;            // 0..31
    const int ty = threadIdx.y;            // 0..7
#pragma unroll
    for (int i = 0; i < 32; i += 8)
        t[ty + i][tx] = W[(size_t)(k0 + ty + i) * N + n0 + tx];
    __syncthreads();
#pragma unroll
    for (int i = 0; i < 32; i += 8)
        Wt[(size_t)(n0 + ty + i) * K + k0 + tx] = f2bf(t[tx][ty + i]);
}

// ---------------------------------------------------------------------------
// LayerNorm (fp32 in) -> bf16 out.  One block (256 threads) per row.
// ---------------------------------------------------------------------------
__global__ void layernorm_bf16_k(const float* __restrict__ in,
                                 const float* __restrict__ w,
                                 const float* __restrict__ b,
                                 unsigned short* __restrict__ out, int D) {
    const int row = blockIdx.x;
    const int tid = threadIdx.x;
    const float* xr = in + (size_t)row * D;

    float s = 0.f, s2 = 0.f;
    for (int i = tid; i < D; i += blockDim.x) {
        float v = xr[i];
        s += v; s2 += v * v;
    }
    for (int off = 16; off; off >>= 1) {
        s  += __shfl_xor(s,  off, 32);
        s2 += __shfl_xor(s2, off, 32);
    }
    __shared__ float rs[8], rs2[8];
    const int lane = tid & 31, wv = tid >> 5;
    if (lane == 0) { rs[wv] = s; rs2[wv] = s2; }
    __syncthreads();
    if (wv == 0) {
        s  = (lane < 8) ? rs[lane]  : 0.f;
        s2 = (lane < 8) ? rs2[lane] : 0.f;
        for (int off = 4; off; off >>= 1) {
            s  += __shfl_xor(s,  off, 32);
            s2 += __shfl_xor(s2, off, 32);
        }
        if (lane == 0) { rs[0] = s; rs2[0] = s2; }
    }
    __syncthreads();
    const float mean = rs[0] / (float)D;
    const float var  = rs2[0] / (float)D - mean * mean;
    const float inv  = rsqrtf(var + 1e-5f);
    unsigned short* orow = out + (size_t)row * D;
    for (int i = tid; i < D; i += blockDim.x)
        orow[i] = f2bf((xr[i] - mean) * inv * w[i] + b[i]);
}

// ---------------------------------------------------------------------------
// GEMM: out[M,N] = EPI( A_bf16[M,K] @ Wt_bf16[N,K]^T + bias[N] )
//   256 threads = 8 waves (4 M x 2 N); block tile 128(M) x 128(N); BK=32.
//   Async global->LDS staging (ASYNCcnt) with LDS double buffering;
//   one barrier per K-step; 8 WMMAs per wave per K-step.
//   EPI: 0 = GELU->bf16, 1 = identity->f32,
//        2 = gate: hres[m,n] += gamma[n]*ybuf[m,n]*sigmoid(v)
// ---------------------------------------------------------------------------
#define LDSA 40                       // padded row stride (elements)
#define TILE_E (128 * LDSA)           // elements per LDS tile
template <int EPI>
__global__ void gemm_bf16_wmma_k(const unsigned short* __restrict__ A,
                                 const unsigned short* __restrict__ Wt,
                                 const float* __restrict__ bias,
                                 unsigned short* __restrict__ outBf,
                                 float* __restrict__ outF,
                                 float* __restrict__ hres,
                                 const float* __restrict__ ybuf,
                                 const float* __restrict__ gamma,
                                 int M, int N, int K) {
    __shared__ unsigned short As[2][TILE_E];   // [buf][m 0..127][k 0..31]
    __shared__ unsigned short Bs[2][TILE_E];   // [buf][n 0..127][k 0..31]

    const int tid  = threadIdx.x;           // 0..255
    const int lane = tid & 31;
    const int wv   = tid >> 5;              // 0..7
    const int wm   = wv >> 1;               // 0..3  (M sub-tile)
    const int wn   = wv & 1;                // 0..1  (N sub-tile)
    const int n0   = blockIdx.x * 128;
    const int m0   = blockIdx.y * 128;

    v8f acc[2][4];
#pragma unroll
    for (int i = 0; i < 2; ++i)
#pragma unroll
        for (int j = 0; j < 4; ++j)
#pragma unroll
            for (int r = 0; r < 8; ++r) acc[i][j][r] = 0.f;

    // staging: each thread owns one 32B (16-element) row chunk of A and of B
    const int sr = tid >> 1;                // row 0..127
    const int sc = (tid & 1) << 4;          // element base 0 or 16
    const unsigned short* pA = A  + (size_t)(m0 + sr) * K + sc;
    const unsigned short* pB = Wt + (size_t)(n0 + sr) * K + sc;
    unsigned ldsA[2], ldsB[2];
#pragma unroll
    for (int b = 0; b < 2; ++b) {
        ldsA[b] = lds_off(&As[b][sr * LDSA + sc]);
        ldsB[b] = lds_off(&Bs[b][sr * LDSA + sc]);
    }

    // fragment indices
    const int l15   = lane & 15;
    const int aoff  = (lane < 16) ? 0 : 16;
    const int boff  = (lane < 16) ? 0 : 16;
    const int arow0 = wm * 32 + l15;
    const int brow0 = wn * 64 + l15;

    const int T = K >> 5;
    // prologue: tile 0 -> buffer 0
    async_copy_b128(ldsA[0],      pA);
    async_copy_b128(ldsA[0] + 16, pA + 8);
    async_copy_b128(ldsB[0],      pB);
    async_copy_b128(ldsB[0] + 16, pB + 8);

    for (int t = 0; t < T; ++t) {
        const int cur = t & 1;
        wait_async0();          // own tile-t chunks landed in LDS
        __syncthreads();        // whole tile t visible; tile t-1 reads done

        if (t + 1 < T) {        // kick off next tile into the other buffer
            const int nxt = cur ^ 1;
            const unsigned short* qA = pA + ((size_t)(t + 1) << 5);
            const unsigned short* qB = pB + ((size_t)(t + 1) << 5);
            async_copy_b128(ldsA[nxt],      qA);
            async_copy_b128(ldsA[nxt] + 16, qA + 8);
            async_copy_b128(ldsB[nxt],      qB);
            async_copy_b128(ldsB[nxt] + 16, qB + 8);
        }

        const unsigned short* as = As[cur];
        const unsigned short* bs = Bs[cur];
        Frag16 af0, af1, bf0, bf1, bf2, bf3;
        af0.u[0] = *(const uint4*)(as + arow0 * LDSA + aoff);
        af0.u[1] = *(const uint4*)(as + arow0 * LDSA + aoff + 8);
        af1.u[0] = *(const uint4*)(as + (arow0 + 16) * LDSA + aoff);
        af1.u[1] = *(const uint4*)(as + (arow0 + 16) * LDSA + aoff + 8);
        bf0.u[0] = *(const uint4*)(bs + (brow0 +  0) * LDSA + boff);
        bf0.u[1] = *(const uint4*)(bs + (brow0 +  0) * LDSA + boff + 8);
        bf1.u[0] = *(const uint4*)(bs + (brow0 + 16) * LDSA + boff);
        bf1.u[1] = *(const uint4*)(bs + (brow0 + 16) * LDSA + boff + 8);
        bf2.u[0] = *(const uint4*)(bs + (brow0 + 32) * LDSA + boff);
        bf2.u[1] = *(const uint4*)(bs + (brow0 + 32) * LDSA + boff + 8);
        bf3.u[0] = *(const uint4*)(bs + (brow0 + 48) * LDSA + boff);
        bf3.u[1] = *(const uint4*)(bs + (brow0 + 48) * LDSA + boff + 8);

        acc[0][0] = __builtin_amdgcn_wmma_f32_16x16x32_bf16(false, af0.v, false, bf0.v, (short)0, acc[0][0], false, false);
        acc[1][0] = __builtin_amdgcn_wmma_f32_16x16x32_bf16(false, af1.v, false, bf0.v, (short)0, acc[1][0], false, false);
        acc[0][1] = __builtin_amdgcn_wmma_f32_16x16x32_bf16(false, af0.v, false, bf1.v, (short)0, acc[0][1], false, false);
        acc[1][1] = __builtin_amdgcn_wmma_f32_16x16x32_bf16(false, af1.v, false, bf1.v, (short)0, acc[1][1], false, false);
        acc[0][2] = __builtin_amdgcn_wmma_f32_16x16x32_bf16(false, af0.v, false, bf2.v, (short)0, acc[0][2], false, false);
        acc[1][2] = __builtin_amdgcn_wmma_f32_16x16x32_bf16(false, af1.v, false, bf2.v, (short)0, acc[1][2], false, false);
        acc[0][3] = __builtin_amdgcn_wmma_f32_16x16x32_bf16(false, af0.v, false, bf3.v, (short)0, acc[0][3], false, false);
        acc[1][3] = __builtin_amdgcn_wmma_f32_16x16x32_bf16(false, af1.v, false, bf3.v, (short)0, acc[1][3], false, false);
    }

    // ---- epilogue (branch-free per template instantiation) ----
    const int mtop = m0 + wm * 32 + ((lane >> 4) << 3);
#pragma unroll
    for (int i = 0; i < 2; ++i) {
#pragma unroll
        for (int j = 0; j < 4; ++j) {
            const int n = n0 + wn * 64 + j * 16 + l15;
            const float bv = bias[n];
            size_t idx = (size_t)(mtop + i * 16) * N + n;
#pragma unroll
            for (int r = 0; r < 8; ++r) {
                const float v = acc[i][j][r] + bv;
                if (EPI == 0) {
                    outBf[idx] = f2bf(gelu_erf(v));
                } else if (EPI == 1) {
                    outF[idx] = v;
                } else {
                    hres[idx] += gamma[n] * ybuf[idx] * sigmoidf_(v);
                }
                idx += (size_t)N;
            }
        }
    }
}

// ---------------------------------------------------------------------------
// Narrow head output: raw[B,outn] = act_bf16[B,hid] @ w2[hid,outn] + b2
// ---------------------------------------------------------------------------
__global__ void head_out_k(const unsigned short* __restrict__ act,
                           const float* __restrict__ w2,
                           const float* __restrict__ b2,
                           float* __restrict__ raw, int hid, int outn) {
    const int row = blockIdx.x;
    const int tid = threadIdx.x;
    float acc[3] = {0.f, 0.f, 0.f};
    const unsigned short* a = act + (size_t)row * hid;
    for (int i = tid; i < hid; i += blockDim.x) {
        const float av = bf2f(a[i]);
        for (int o = 0; o < outn; ++o) acc[o] += av * w2[(size_t)i * outn + o];
    }
    __shared__ float red[256];
    for (int o = 0; o < outn; ++o) {
        red[tid] = acc[o];
        __syncthreads();
        for (int st = 128; st; st >>= 1) {
            if (tid < st) red[tid] += red[tid + st];
            __syncthreads();
        }
        if (tid == 0) raw[(size_t)row * outn + o] = red[0] + b2[o];
        __syncthreads();
    }
}

// ---------------------------------------------------------------------------
// Final combine
// ---------------------------------------------------------------------------
__global__ void combine_k(const float* __restrict__ t_raw, const float* __restrict__ r_raw,
                          const float* __restrict__ g_raw, const float* __restrict__ s_raw,
                          float* __restrict__ out, int B) {
    const int i = blockIdx.x * blockDim.x + threadIdx.x;
    if (i >= B) return;
    const float s = 0.5f + 0.5f * sigmoidf_(s_raw[i]);
    const float g = 2.0f * sigmoidf_(g_raw[i]) - 1.0f;
    float o[7];
    o[0] = tanhf(t_raw[i * 3 + 0]) * 0.05f;
    o[1] = tanhf(t_raw[i * 3 + 1]) * 0.05f;
    o[2] = tanhf(t_raw[i * 3 + 2]) * 0.05f;
    o[3] = tanhf(r_raw[i * 3 + 0]) * 0.25f;
    o[4] = tanhf(r_raw[i * 3 + 1]) * 0.25f;
    o[5] = tanhf(r_raw[i * 3 + 2]) * 0.25f;
    o[6] = g;
#pragma unroll
    for (int k = 0; k < 7; ++k) out[(size_t)i * 7 + k] = o[k] * s;
}

// ---------------------------------------------------------------------------
// Host orchestration
// ---------------------------------------------------------------------------
static inline void conv_weight(const float* W, unsigned short* Wt, int K, int N,
                               hipStream_t stream) {
    convT_bf16_k<<<dim3(N / 32, K / 32), dim3(32, 8), 0, stream>>>(W, Wt, K, N);
}

extern "C" void kernel_launch(void* const* d_in, const int* in_sizes, int n_in,
                              void* d_out, int out_size, void* d_ws, size_t ws_size,
                              hipStream_t stream) {
    (void)in_sizes; (void)n_in; (void)out_size; (void)ws_size;
    const int Bn = 4096, Dd = 2048, Hh = 8192, SEn = 512, Ll = 4;

    // ---- inputs (setup_inputs dict order, depth-first) ----
    const float* x        = (const float*)d_in[0];
    const float* token    = (const float*)d_in[1];
    const float* t_ln_w   = (const float*)d_in[2];
    const float* t_ln_b   = (const float*)d_in[3];
    const float* fc1_w    = (const float*)d_in[4];
    const float* fc1_b    = (const float*)d_in[5];
    const float* fc2_w    = (const float*)d_in[6];
    const float* fc2_b    = (const float*)d_in[7];
    const float* se_ln_w  = (const float*)d_in[8];
    const float* se_ln_b  = (const float*)d_in[9];
    const float* se1_w    = (const float*)d_in[10];
    const float* se1_b    = (const float*)d_in[11];
    const float* se2_w    = (const float*)d_in[12];
    const float* se2_b    = (const float*)d_in[13];
    const float* gamma    = (const float*)d_in[14];
    const float* const* H_TR = (const float* const*)&d_in[15];
    const float* const* H_RO = (const float* const*)&d_in[21];
    const float* const* H_GR = (const float* const*)&d_in[27];
    const float* const* H_ST = (const float* const*)&d_in[33];

    // ---- workspace carve-out ----
    char* ws = (char*)d_ws;
    float*          h     = (float*)ws;          ws += (size_t)Bn * Dd * 4;
    float*          ybuf  = (float*)ws;          ws += (size_t)Bn * Dd * 4;
    unsigned short* xbf   = (unsigned short*)ws; ws += (size_t)Bn * Dd * 2;
    unsigned short* actbf = (unsigned short*)ws; ws += (size_t)Bn * Hh * 2;
    unsigned short* sebf  = (unsigned short*)ws; ws += (size_t)Bn * SEn * 2;
    unsigned short* wT    = (unsigned short*)ws; ws += (size_t)Dd * Hh * 2;  // largest weight
    float*          t_raw = (float*)ws;          ws += (size_t)Bn * 3 * 4;
    float*          r_raw = (float*)ws;          ws += (size_t)Bn * 3 * 4;
    float*          g_raw = (float*)ws;          ws += (size_t)Bn * 4;
    float*          s_raw = (float*)ws;          ws += (size_t)Bn * 4;

    const dim3 blk256(256);

    // h = x + task_token
    init_h_k<<<(Bn * Dd + 255) / 256, blk256, 0, stream>>>(x, token, h, Bn * Dd, Dd);

    // ---- trunk: 4 residual SE-MLP blocks ----
    for (int l = 0; l < Ll; ++l) {
        const float* l_ln_w = t_ln_w  + (size_t)l * Dd;
        const float* l_ln_b = t_ln_b  + (size_t)l * Dd;
        const float* l_f1w  = fc1_w   + (size_t)l * Dd * Hh;
        const float* l_f1b  = fc1_b   + (size_t)l * Hh;
        const float* l_f2w  = fc2_w   + (size_t)l * Hh * Dd;
        const float* l_f2b  = fc2_b   + (size_t)l * Dd;
        const float* l_slnw = se_ln_w + (size_t)l * Dd;
        const float* l_slnb = se_ln_b + (size_t)l * Dd;
        const float* l_s1w  = se1_w   + (size_t)l * Dd * SEn;
        const float* l_s1b  = se1_b   + (size_t)l * SEn;
        const float* l_s2w  = se2_w   + (size_t)l * SEn * Dd;
        const float* l_s2b  = se2_b   + (size_t)l * Dd;
        const float* l_gam  = gamma   + (size_t)l * Dd;

        layernorm_bf16_k<<<Bn, blk256, 0, stream>>>(h, l_ln_w, l_ln_b, xbf, Dd);
        conv_weight(l_f1w, wT, Dd, Hh, stream);
        gemm_bf16_wmma_k<0><<<dim3(Hh / 128, Bn / 128), blk256, 0, stream>>>(
            xbf, wT, l_f1b, actbf, nullptr, nullptr, nullptr, nullptr, Bn, Hh, Dd);
        conv_weight(l_f2w, wT, Hh, Dd, stream);
        gemm_bf16_wmma_k<1><<<dim3(Dd / 128, Bn / 128), blk256, 0, stream>>>(
            actbf, wT, l_f2b, nullptr, ybuf, nullptr, nullptr, nullptr, Bn, Dd, Hh);
        layernorm_bf16_k<<<Bn, blk256, 0, stream>>>(ybuf, l_slnw, l_slnb, xbf, Dd);
        conv_weight(l_s1w, wT, Dd, SEn, stream);
        gemm_bf16_wmma_k<0><<<dim3(SEn / 128, Bn / 128), blk256, 0, stream>>>(
            xbf, wT, l_s1b, sebf, nullptr, nullptr, nullptr, nullptr, Bn, SEn, Dd);
        conv_weight(l_s2w, wT, SEn, Dd, stream);
        gemm_bf16_wmma_k<2><<<dim3(Dd / 128, Bn / 128), blk256, 0, stream>>>(
            sebf, wT, l_s2b, nullptr, nullptr, h, ybuf, l_gam, Bn, Dd, SEn);
    }

    // ---- heads ----
    struct HeadDesc { const float* const* p; int hid; int outn; float* raw; };
    HeadDesc heads[4] = {
        { H_TR, Dd,     3, t_raw },
        { H_RO, Dd,     3, r_raw },
        { H_GR, Dd / 2, 1, g_raw },
        { H_ST, Dd / 2, 1, s_raw },
    };
    for (int hd = 0; hd < 4; ++hd) {
        const float* ln_w = heads[hd].p[0];
        const float* ln_b = heads[hd].p[1];
        const float* w1   = heads[hd].p[2];
        const float* b1   = heads[hd].p[3];
        const float* w2   = heads[hd].p[4];
        const float* b2   = heads[hd].p[5];
        const int hid = heads[hd].hid;

        layernorm_bf16_k<<<Bn, blk256, 0, stream>>>(h, ln_w, ln_b, xbf, Dd);
        conv_weight(w1, wT, Dd, hid, stream);
        gemm_bf16_wmma_k<0><<<dim3(hid / 128, Bn / 128), blk256, 0, stream>>>(
            xbf, wT, b1, actbf, nullptr, nullptr, nullptr, nullptr, Bn, hid, Dd);
        head_out_k<<<Bn, blk256, 0, stream>>>(actbf, w2, b2, heads[hd].raw, hid, heads[hd].outn);
    }

    combine_k<<<(Bn + 255) / 256, blk256, 0, stream>>>(t_raw, r_raw, g_raw, s_raw,
                                                       (float*)d_out, Bn);
}
